// FourLayer_64F_25898652795643
// MI455X (gfx1250) — compile-verified, matrix operand order checked by reading, not run on hardware
//
#include <hip/hip_runtime.h>
#include <hip/hip_bf16.h>
#include <math.h>

// ---------------------------------------------------------------------------
// MI455X (gfx1250) implementation.
// Heavy ops (conv-as-implicit-GEMM, cosine-similarity GEMM) use
// v_wmma_f32_16x16x32_f16 (f16 operands, f32 accumulate). Glue ops are scalar.
// ---------------------------------------------------------------------------

typedef __attribute__((ext_vector_type(16))) _Float16 v16h;
typedef __attribute__((ext_vector_type(8)))  float    v8f;

#define LRELU(x) ((x) > 0.f ? (x) : 0.2f * (x))

union FragU { v16h h; unsigned u[8]; };

// Load one WMMA 16-bit fragment for a lane whose matrix row (A) / column (B)
// data lives contiguously along K at `row`. Packed-pair K mapping per ISA
// 7.12.2: VGPR v (v<4): K = 2v + {0,1} + 8*half ; (v>=4): K = 8 + 2v + {0,1} + 8*half.
__device__ __forceinline__ v16h load_frag_row(const _Float16* __restrict__ row,
                                              int k0, int half) {
  FragU f;
#pragma unroll
  for (int v = 0; v < 8; ++v) {
    int kk = k0 + ((v < 4) ? (2 * v) : (8 + 2 * v)) + 8 * half;
    f.u[v] = *(const unsigned*)(row + kk);
  }
  return f.h;
}

// ---------------------------------------------------------------------------
// Implicit-GEMM conv core: D[64][Mpad] = W16[64][Kpad] x P[Mpad][Kpad]^T
// Block = 128 threads = 4 waves; wave w handles output channels [16w,16w+16).
// blockIdx.x = M tile (16 positions).
// ---------------------------------------------------------------------------
__global__ void k_conv_gemm(const _Float16* __restrict__ W16,
                            const _Float16* __restrict__ P,
                            float* __restrict__ Out,
                            int Kpad, int Mpad) {
  int lane  = threadIdx.x & 31;
  int wave  = threadIdx.x >> 5;
  int row16 = lane & 15;
  int half  = lane >> 4;
  const _Float16* arow = W16 + (size_t)(wave * 16 + row16) * Kpad;
  const _Float16* brow = P + ((size_t)blockIdx.x * 16 + row16) * (size_t)Kpad;
  v8f acc = {};
  for (int k0 = 0; k0 < Kpad; k0 += 32) {
    if (k0 + 32 < Kpad) __builtin_prefetch(brow + k0 + 32, 0, 1);
    v16h a = load_frag_row(arow, k0, half);
    v16h b = load_frag_row(brow, k0, half);
    acc = __builtin_amdgcn_wmma_f32_16x16x32_f16(false, a, false, b,
                                                 (short)0, acc, false, false);
  }
#pragma unroll
  for (int r = 0; r < 8; ++r) {
    int cout = wave * 16 + r + 8 * half;          // D: M = r + 8*half
    size_t m = (size_t)blockIdx.x * 16 + row16;   // D: N = lane%16
    Out[(size_t)cout * Mpad + m] = acc[r];
  }
}

// im2col: activations ([C][N][H][W] channel-major, or NCHW for the raw input)
// -> f16 patch matrix P[Mpad][Kpad], zero padded in both M and K.
__global__ void k_im2col(const float* __restrict__ X, _Float16* __restrict__ P,
                         int N, int Cin, int H, int W, int OH, int OW,
                         int stride, int pad, int Kpad, int Mpad,
                         long long strideC, int nchw) {
  size_t idx = (size_t)blockIdx.x * blockDim.x + threadIdx.x;
  size_t total = (size_t)Mpad * Kpad;
  if (idx >= total) return;
  int k = (int)(idx % Kpad);
  size_t m = idx / Kpad;
  float val = 0.f;
  int M = N * OH * OW;
  int Kr = Cin * 9;
  if (m < (size_t)M && k < Kr) {
    int ci = k / 9, rr = k - ci * 9, ky = rr / 3, kx = rr - ky * 3;
    int n = (int)(m / (OH * OW));
    int r2 = (int)(m % (OH * OW));
    int oy = r2 / OW, ox = r2 - oy * OW;
    int iy = oy * stride - pad + ky, ix = ox * stride - pad + kx;
    if (iy >= 0 && iy < H && ix >= 0 && ix < W) {
      size_t src = nchw ? ((size_t)(n * Cin + ci) * H + iy) * W + ix
                        : (size_t)ci * strideC + ((size_t)n * H + iy) * W + ix;
      val = X[src];
    }
  }
  P[idx] = (_Float16)val;
}

// Pack f32 conv weights (Cout=64, Cin, 3, 3) into f16 [64][Kpad].
__global__ void k_pack_w(const float* __restrict__ w, _Float16* __restrict__ W16,
                         int K, int Kpad) {
  int idx = blockIdx.x * blockDim.x + threadIdx.x;
  if (idx >= 64 * Kpad) return;
  int co = idx / Kpad, k = idx - co * Kpad;
  W16[idx] = (_Float16)((k < K) ? w[co * K + k] : 0.f);
}

// Per-channel batch statistics over m < M (stride Mstride).
__global__ void k_bn_stats(const float* __restrict__ X, float* __restrict__ mean,
                           float* __restrict__ var, int Mstride, int M) {
  __shared__ float ss[256], sq[256];
  int c = blockIdx.x, t = threadIdx.x;
  const float* xc = X + (size_t)c * Mstride;
  float s = 0.f, q = 0.f;
  for (int m = t; m < M; m += 256) { float v = xc[m]; s += v; q += v * v; }
  ss[t] = s; sq[t] = q;
  __syncthreads();
  for (int o = 128; o > 0; o >>= 1) {
    if (t < o) { ss[t] += ss[t + o]; sq[t] += sq[t + o]; }
    __syncthreads();
  }
  if (t == 0) {
    float mu = ss[0] / M;
    mean[c] = mu;
    var[c] = sq[0] / M - mu * mu;
  }
}

__global__ void k_bn_lrelu(float* __restrict__ X, const float* __restrict__ mean,
                           const float* __restrict__ var, const float* __restrict__ g,
                           const float* __restrict__ b, int Mstride, int M) {
  size_t idx = (size_t)blockIdx.x * blockDim.x + threadIdx.x;
  if (idx >= (size_t)64 * M) return;
  int c = (int)(idx / M);
  int m = (int)(idx - (size_t)c * M);
  float* p = X + (size_t)c * Mstride + m;
  float y = (*p - mean[c]) * rsqrtf(var[c] + 1e-5f) * g[c] + b[c];
  *p = LRELU(y);
}

__global__ void k_lrelu(float* __restrict__ X, int n) {
  int i = blockIdx.x * blockDim.x + threadIdx.x;
  if (i < n) { float v = X[i]; X[i] = LRELU(v); }
}

// Max-pool on channel-major activations.
__global__ void k_maxpool(const float* __restrict__ in, float* __restrict__ out,
                          int N, int H, int W, int OH, int OW, int k, int s,
                          long long sCin, long long sCout) {
  size_t idx = (size_t)blockIdx.x * blockDim.x + threadIdx.x;
  size_t total = (size_t)64 * N * OH * OW;
  if (idx >= total) return;
  int per = N * OH * OW;
  int c = (int)(idx / per);
  int r = (int)(idx - (size_t)c * per);
  int n = r / (OH * OW);
  int r2 = r - n * (OH * OW);
  int oy = r2 / OW, ox = r2 - oy * OW;
  const float* base = in + (size_t)c * sCin + (size_t)n * H * W;
  float mv = -INFINITY;
  for (int ky = 0; ky < k; ++ky)
    for (int kx = 0; kx < k; ++kx)
      mv = fmaxf(mv, base[(oy * s + ky) * W + ox * s + kx]);
  out[(size_t)c * sCout + (size_t)n * OH * OW + r2] = mv;
}

// ConvTranspose2d(64->64, k=4, s=1), input 1x1: out[co][n][yx] = sum_ci h[ci][n]*w[ci][co][yx]
__global__ void k_deconv1(const float* __restrict__ h, const float* __restrict__ w,
                          float* __restrict__ out, int N) {
  int idx = blockIdx.x * blockDim.x + threadIdx.x;
  if (idx >= 64 * N * 16) return;
  int co = idx / (N * 16);
  int r = idx - co * (N * 16);
  int n = r / 16, yx = r - n * 16;
  float acc = 0.f;
  for (int ci = 0; ci < 64; ++ci)
    acc += h[ci * 32 + n] * w[(ci * 64 + co) * 16 + yx];
  out[co * (N * 16) + r] = acc;
}

// ConvTranspose2d(64->64, k=4, s=2), 4x4 -> 10x10
__global__ void k_deconv2(const float* __restrict__ in, const float* __restrict__ w,
                          float* __restrict__ out, int N) {
  int idx = blockIdx.x * blockDim.x + threadIdx.x;
  if (idx >= 64 * N * 100) return;
  int co = idx / (N * 100);
  int r = idx - co * (N * 100);
  int n = r / 100, p = r - n * 100;
  int oy = p / 10, ox = p - oy * 10;
  float acc = 0.f;
  for (int ky = 0; ky < 4; ++ky) {
    int ty = oy - ky;
    if (ty < 0 || (ty & 1) || (ty >> 1) >= 4) continue;
    int iy = ty >> 1;
    for (int kx = 0; kx < 4; ++kx) {
      int tx = ox - kx;
      if (tx < 0 || (tx & 1) || (tx >> 1) >= 4) continue;
      int ix = tx >> 1;
      for (int ci = 0; ci < 64; ++ci)
        acc += in[ci * (N * 16) + n * 16 + iy * 4 + ix] * w[(ci * 64 + co) * 16 + ky * 4 + kx];
    }
  }
  out[co * (N * 100) + r] = acc;
}

// ConvTranspose2d(64->1, k=3, s=2), 10x10 -> 21x21
__global__ void k_deconv3(const float* __restrict__ in, const float* __restrict__ w,
                          float* __restrict__ out, int N) {
  int idx = blockIdx.x * blockDim.x + threadIdx.x;
  if (idx >= N * 441) return;
  int n = idx / 441, p = idx - n * 441;
  int oy = p / 21, ox = p - oy * 21;
  float acc = 0.f;
  for (int ky = 0; ky < 3; ++ky) {
    int ty = oy - ky;
    if (ty < 0 || (ty & 1) || (ty >> 1) >= 10) continue;
    int iy = ty >> 1;
    for (int kx = 0; kx < 3; ++kx) {
      int tx = ox - kx;
      if (tx < 0 || (tx & 1) || (tx >> 1) >= 10) continue;
      int ix = tx >> 1;
      for (int ci = 0; ci < 64; ++ci)
        acc += in[ci * (N * 100) + n * 100 + iy * 10 + ix] * w[ci * 9 + ky * 3 + kx];
    }
  }
  out[idx] = acc;
}

// Per-image: argmax of deconv map -> mi; linear heads -> sigma inverse + coeff.
// (softmax before argmax is monotone -> argmax of logits)
__global__ void k_mi_sigma(const float* __restrict__ D3, const float* __restrict__ H,
                           const float* __restrict__ Wx, const float* __restrict__ bx,
                           const float* __restrict__ Wy, const float* __restrict__ by,
                           const float* __restrict__ Wt, const float* __restrict__ bt,
                           float* __restrict__ par, int N) {
  int n = blockIdx.x * blockDim.x + threadIdx.x;
  if (n >= N) return;
  const float* d = D3 + n * 441;
  int best = 0; float bv = d[0];
  for (int j = 1; j < 441; ++j) { float v = d[j]; if (v > bv) { bv = v; best = j; } }
  float mi0 = (float)(best / 21 - 10) * 0.1f;
  float mi1 = (float)(best % 21 - 10) * 0.1f;
  float f[64];
  for (int c = 0; c < 64; ++c) f[c] = H[c * 32 + n];
  int jx = 0, jy = 0, jt = 0;
  float blx = -1e30f, bly = -1e30f, blt = -1e30f;
  for (int j = 0; j < 6; ++j) {
    float lx = bx[j], ly = by[j], lt = bt[j];
    for (int c = 0; c < 64; ++c) {
      lx += f[c] * Wx[j * 64 + c];
      ly += f[c] * Wy[j * 64 + c];
      lt += f[c] * Wt[j * 64 + c];
    }
    if (lx > blx) { blx = lx; jx = j; }
    if (ly > bly) { bly = ly; jy = j; }
    if (lt > blt) { blt = lt; jt = j; }
  }
  float xs = jx / 5.0f, ys = jy / 5.0f, th = jt * 0.5235987755982988f;
  float a = xs * 0.25f + 0.1f, bq = ys * 0.25f + 0.15f;
  float cs = cosf(th), sn = sinf(th);
  float s00 = cs * cs * a + sn * sn * bq;
  float s01 = cs * sn * (a - bq);
  float s11 = sn * sn * a + cs * cs * bq;
  float det = a * bq;                       // rotation preserves determinant
  float* pp = par + n * 6;
  pp[0] = mi0; pp[1] = mi1;
  pp[2] = s11 / det; pp[3] = -s01 / det; pp[4] = s00 / det;
  pp[5] = 1.f / (6.283185307179586f * sqrtf(det));
}

__global__ void k_mask(const float* __restrict__ par, float* __restrict__ mask,
                       int N, int imgOff) {
  int idx = blockIdx.x * blockDim.x + threadIdx.x;
  if (idx >= N * 441) return;
  int n = idx / 441, p = idx - n * 441;
  const float* pp = par + n * 6;
  int i = p / 21, j = p - i * 21;
  float dx = (float)(i - 10) * 0.1f - pp[0];
  float dy = (float)(j - 10) * 0.1f - pp[1];
  float q = pp[2] * dx * dx + 2.f * pp[3] * dx * dy + pp[4] * dy * dy;
  mask[(size_t)(imgOff + n) * 441 + p] = expf(-0.5f * q) * pp[5];
}

// L2-normalize the 64-channel feature vector per (image, position) -> f16 rows.
__global__ void k_normalize(const float* __restrict__ A4, _Float16* __restrict__ F,
                            int Mstride, int N, int imgOff) {
  int idx = blockIdx.x * blockDim.x + threadIdx.x;
  if (idx >= N * 441) return;
  int n = idx / 441, p = idx - n * 441;
  size_t m = (size_t)n * 441 + p;
  float s = 0.f;
  for (int c = 0; c < 64; ++c) { float v = A4[(size_t)c * Mstride + m]; s += v * v; }
  float inv = rsqrtf(s);
  _Float16* out = F + ((size_t)(imgOff + n) * 441 + p) * 64;
  for (int c = 0; c < 64; ++c)
    out[c] = (_Float16)(A4[(size_t)c * Mstride + m] * inv);
}

// Fused similarity GEMM (WMMA, K=64) + masking + per-row running top-3 + tile sum.
// grid = (28 m-tiles, K=5, B=25); block = 1 wave (32).
__global__ void k_sims_topk(const _Float16* __restrict__ F,
                            const float* __restrict__ mask,
                            float* __restrict__ partial) {
  const int NMT = 28;
  int mt = blockIdx.x, kk = blockIdx.y, b = blockIdx.z;
  int lane = threadIdx.x & 31;
  int row16 = lane & 15, half = lane >> 4;
  int qpos = mt * 16 + row16;
  int qc = (qpos < 441) ? qpos : 440;
  const _Float16* arow = F + ((size_t)b * 441 + qc) * 64;
  v16h a0 = load_frag_row(arow, 0, half);
  v16h a1 = load_frag_row(arow, 32, half);
  float m1 = (qpos < 441) ? mask[b * 441 + qpos] : 0.f;
  __shared__ float tile[16][17];
  __shared__ float m2s[16];
  __shared__ float red[32];
  float t0 = -1e30f, t1 = -1e30f, t2 = -1e30f;
  for (int nt = 0; nt < 138; ++nt) {
    int scol = nt * 16 + row16;
    int sc = (scol < 2205) ? scol : 0;
    int simg = 25 + kk * 5 + sc / 441;
    int spos = sc % 441;
    const _Float16* brow = F + ((size_t)simg * 441 + spos) * 64;
    v16h b0 = load_frag_row(brow, 0, half);
    v16h b1 = load_frag_row(brow, 32, half);
    v8f acc = {};
    acc = __builtin_amdgcn_wmma_f32_16x16x32_f16(false, a0, false, b0,
                                                 (short)0, acc, false, false);
    acc = __builtin_amdgcn_wmma_f32_16x16x32_f16(false, a1, false, b1,
                                                 (short)0, acc, false, false);
    if (half == 0)
      m2s[row16] = (scol < 2205) ? mask[(size_t)simg * 441 + spos] : 0.f;
#pragma unroll
    for (int r = 0; r < 8; ++r) tile[r + 8 * half][row16] = acc[r];
    __syncthreads();
    if (lane < 16) {
      int cmax = 2205 - nt * 16; if (cmax > 16) cmax = 16;
      const float* tr = tile[lane];
      for (int j = 0; j < cmax; ++j) {
        float v = tr[j] * m1 * m2s[j];
        if (v > t0)      { t2 = t1; t1 = t0; t0 = v; }
        else if (v > t1) { t2 = t1; t1 = v; }
        else if (v > t2) { t2 = v; }
      }
    }
    __syncthreads();
  }
  float s = 0.f;
  if (lane < 16 && (mt * 16 + lane) < 441) s = t0 + t1 + t2;
  red[lane] = s;
  __syncthreads();
  if (lane == 0) {
    float acc = 0.f;
    for (int i = 0; i < 16; ++i) acc += red[i];
    partial[((size_t)b * 5 + kk) * NMT + mt] = acc;
  }
}

__global__ void k_reduce(const float* __restrict__ partial, float* __restrict__ out) {
  int i = blockIdx.x * blockDim.x + threadIdx.x;
  if (i >= 125) return;
  float s = 0.f;
  for (int t = 0; t < 28; ++t) s += partial[i * 28 + t];
  out[i] = s;
}

// ---------------------------------------------------------------------------
static inline int gb(size_t n, int b) { return (int)((n + b - 1) / b); }

extern "C" void kernel_launch(void* const* d_in, const int* in_sizes, int n_in,
                              void* d_out, int out_size, void* d_ws, size_t ws_size,
                              hipStream_t stream) {
  (void)in_sizes; (void)n_in; (void)out_size; (void)ws_size;
  // setup_inputs() insertion order: input1, input2, then params in insertion order.
  const float* input1 = (const float*)d_in[0];
  const float* input2 = (const float*)d_in[1];
  const float* w1  = (const float*)d_in[2];  const float* g1  = (const float*)d_in[3];  const float* b1  = (const float*)d_in[4];
  const float* w2  = (const float*)d_in[5];  const float* g2  = (const float*)d_in[6];  const float* b2  = (const float*)d_in[7];
  const float* w3  = (const float*)d_in[8];  const float* g3  = (const float*)d_in[9];  const float* b3  = (const float*)d_in[10];
  const float* w4  = (const float*)d_in[11]; const float* g4  = (const float*)d_in[12]; const float* b4  = (const float*)d_in[13];
  const float* gw1 = (const float*)d_in[14]; const float* gg1 = (const float*)d_in[15]; const float* gb1 = (const float*)d_in[16];
  const float* gw2 = (const float*)d_in[17]; const float* gg2 = (const float*)d_in[18]; const float* gb2 = (const float*)d_in[19];
  const float* gw3 = (const float*)d_in[20];
  const float* dw1 = (const float*)d_in[21]; const float* dg1 = (const float*)d_in[22]; const float* db1 = (const float*)d_in[23];
  const float* dw2 = (const float*)d_in[24]; const float* dg2 = (const float*)d_in[25]; const float* db2 = (const float*)d_in[26];
  const float* dw3 = (const float*)d_in[27];
  const float* Wx = (const float*)d_in[28]; const float* bx = (const float*)d_in[29];
  const float* Wy = (const float*)d_in[30]; const float* by = (const float*)d_in[31];
  const float* Wt = (const float*)d_in[32]; const float* bt = (const float*)d_in[33];

  char* ws = (char*)d_ws;
  size_t off = 0;
  auto carve = [&](size_t bytes) -> void* {
    off = (off + 255) & ~(size_t)255;
    void* p = ws + off;
    off += bytes;
    return p;
  };

  _Float16* W16[7];
  W16[0] = (_Float16*)carve((size_t)64 * 32 * 2);
  for (int i = 1; i < 7; ++i) W16[i] = (_Float16*)carve((size_t)64 * 576 * 2);
  _Float16* IM  = (_Float16*)carve((size_t)44112 * 576 * 2);  // largest im2col
  float* CO  = (float*)carve((size_t)64 * 176400 * 4);        // conv outputs
  float* PO  = (float*)carve((size_t)64 * 44100 * 4);         // pooled acts
  float* A4  = (float*)carve((size_t)64 * 11040 * 4);         // features output
  float* GP  = (float*)carve((size_t)64 * 1225 * 4);          // pooled gw1 (7x7)
  float* HB  = (float*)carve((size_t)64 * 32 * 4);            // gaussian h
  float* D1  = (float*)carve((size_t)64 * 400 * 4);
  float* D2  = (float*)carve((size_t)64 * 2500 * 4);
  float* D3  = (float*)carve((size_t)25 * 441 * 4);
  float* PAR = (float*)carve((size_t)25 * 6 * 4);
  float* MEAN = (float*)carve(64 * 4);
  float* VAR  = (float*)carve(64 * 4);
  float* MASK = (float*)carve((size_t)50 * 441 * 4);
  _Float16* FN = (_Float16*)carve((size_t)50 * 441 * 64 * 2);
  float* PARTIAL = (float*)carve((size_t)3500 * 4);

  // Pack all conv weights to f16 once.
  k_pack_w<<<gb(64 * 32, 256), 256, 0, stream>>>(w1, W16[0], 27, 32);
  const float* cw[6] = { w2, w3, w4, gw1, gw2, gw3 };
  for (int i = 0; i < 6; ++i)
    k_pack_w<<<gb(64 * 576, 256), 256, 0, stream>>>(cw[i], W16[i + 1], 576, 576);

  // The reference runs features()/gaussian() separately on input1 and input2,
  // so batch-norm statistics differ -> run the full pipeline twice (N=25 each).
  for (int call = 0; call < 2; ++call) {
    const float* X = call ? input2 : input1;
    int imgOff = call * 25;

    // conv1: 3->64, 84x84, pad1.  M=176400, Kpad=32.
    k_im2col<<<gb((size_t)176400 * 32, 256), 256, 0, stream>>>(X, IM, 25, 3, 84, 84, 84, 84, 1, 1, 32, 176400, 0, 1);
    k_conv_gemm<<<176400 / 16, 128, 0, stream>>>(W16[0], IM, CO, 32, 176400);
    k_bn_stats<<<64, 256, 0, stream>>>(CO, MEAN, VAR, 176400, 176400);
    k_bn_lrelu<<<gb((size_t)64 * 176400, 256), 256, 0, stream>>>(CO, MEAN, VAR, g1, b1, 176400, 176400);
    k_maxpool<<<gb((size_t)64 * 25 * 42 * 42, 256), 256, 0, stream>>>(CO, PO, 25, 84, 84, 42, 42, 2, 2, 176400, 44100);

    // conv2: 64->64, 42x42, pad1.  M=44100, Mpad=44112.
    k_im2col<<<gb((size_t)44112 * 576, 256), 256, 0, stream>>>(PO, IM, 25, 64, 42, 42, 42, 42, 1, 1, 576, 44112, 44100, 0);
    k_conv_gemm<<<44112 / 16, 128, 0, stream>>>(W16[1], IM, CO, 576, 44112);
    k_bn_stats<<<64, 256, 0, stream>>>(CO, MEAN, VAR, 44112, 44100);
    k_bn_lrelu<<<gb((size_t)64 * 44100, 256), 256, 0, stream>>>(CO, MEAN, VAR, g2, b2, 44112, 44100);
    k_maxpool<<<gb((size_t)64 * 25 * 21 * 21, 256), 256, 0, stream>>>(CO, PO, 25, 42, 42, 21, 21, 2, 2, 44112, 11025);

    // conv3: 64->64, 21x21, pad1.  M=11025, Mpad=11040.
    k_im2col<<<gb((size_t)11040 * 576, 256), 256, 0, stream>>>(PO, IM, 25, 64, 21, 21, 21, 21, 1, 1, 576, 11040, 11025, 0);
    k_conv_gemm<<<11040 / 16, 128, 0, stream>>>(W16[2], IM, CO, 576, 11040);
    k_bn_stats<<<64, 256, 0, stream>>>(CO, MEAN, VAR, 11040, 11025);
    k_bn_lrelu<<<gb((size_t)64 * 11025, 256), 256, 0, stream>>>(CO, MEAN, VAR, g3, b3, 11040, 11025);

    // conv4: 64->64, 21x21, pad1 -> A4 (features output).
    k_im2col<<<gb((size_t)11040 * 576, 256), 256, 0, stream>>>(CO, IM, 25, 64, 21, 21, 21, 21, 1, 1, 576, 11040, 11040, 0);
    k_conv_gemm<<<11040 / 16, 128, 0, stream>>>(W16[3], IM, A4, 576, 11040);
    k_bn_stats<<<64, 256, 0, stream>>>(A4, MEAN, VAR, 11040, 11025);
    k_bn_lrelu<<<gb((size_t)64 * 11025, 256), 256, 0, stream>>>(A4, MEAN, VAR, g4, b4, 11040, 11025);

    // gw1: 64->64 pad1 + maxpool3 (21 -> 7).
    k_im2col<<<gb((size_t)11040 * 576, 256), 256, 0, stream>>>(A4, IM, 25, 64, 21, 21, 21, 21, 1, 1, 576, 11040, 11040, 0);
    k_conv_gemm<<<11040 / 16, 128, 0, stream>>>(W16[4], IM, CO, 576, 11040);
    k_bn_stats<<<64, 256, 0, stream>>>(CO, MEAN, VAR, 11040, 11025);
    k_bn_lrelu<<<gb((size_t)64 * 11025, 256), 256, 0, stream>>>(CO, MEAN, VAR, gg1, gb1, 11040, 11025);
    k_maxpool<<<gb((size_t)64 * 25 * 49, 256), 256, 0, stream>>>(CO, GP, 25, 21, 21, 7, 7, 3, 3, 11040, 1225);

    // gw2: 64->64, stride2 pad0 (7 -> 3).  M=225, Mpad=240.
    k_im2col<<<gb((size_t)240 * 576, 256), 256, 0, stream>>>(GP, IM, 25, 64, 7, 7, 3, 3, 2, 0, 576, 240, 1225, 0);
    k_conv_gemm<<<240 / 16, 128, 0, stream>>>(W16[5], IM, CO, 576, 240);
    k_bn_stats<<<64, 256, 0, stream>>>(CO, MEAN, VAR, 240, 225);
    k_bn_lrelu<<<gb((size_t)64 * 225, 256), 256, 0, stream>>>(CO, MEAN, VAR, gg2, gb2, 240, 225);

    // gw3: 64->64, pad0 (3 -> 1), lrelu only.  M=25, Mpad=32.
    k_im2col<<<gb((size_t)32 * 576, 256), 256, 0, stream>>>(CO, IM, 25, 64, 3, 3, 1, 1, 1, 0, 576, 32, 240, 0);
    k_conv_gemm<<<32 / 16, 128, 0, stream>>>(W16[6], IM, HB, 576, 32);
    k_lrelu<<<gb(64 * 32, 256), 256, 0, stream>>>(HB, 64 * 32);

    // Deconv / head branch (tiny).
    k_deconv1<<<gb((size_t)64 * 25 * 16, 256), 256, 0, stream>>>(HB, dw1, D1, 25);
    k_bn_stats<<<64, 256, 0, stream>>>(D1, MEAN, VAR, 400, 400);
    k_bn_lrelu<<<gb((size_t)64 * 400, 256), 256, 0, stream>>>(D1, MEAN, VAR, dg1, db1, 400, 400);
    k_deconv2<<<gb((size_t)64 * 25 * 100, 256), 256, 0, stream>>>(D1, dw2, D2, 25);
    k_bn_stats<<<64, 256, 0, stream>>>(D2, MEAN, VAR, 2500, 2500);
    k_bn_lrelu<<<gb((size_t)64 * 2500, 256), 256, 0, stream>>>(D2, MEAN, VAR, dg2, db2, 2500, 2500);
    k_deconv3<<<gb((size_t)25 * 441, 256), 256, 0, stream>>>(D2, dw3, D3, 25);
    k_mi_sigma<<<1, 32, 0, stream>>>(D3, HB, Wx, bx, Wy, by, Wt, bt, PAR, 25);
    k_mask<<<gb((size_t)25 * 441, 256), 256, 0, stream>>>(PAR, MASK, 25, imgOff);

    // Normalized f16 feature rows for the similarity GEMM.
    k_normalize<<<gb((size_t)25 * 441, 256), 256, 0, stream>>>(A4, FN, 11040, 25, imgOff);
  }

  // Fused cosine-similarity GEMM + mask + top-3, then reduce to (25,5).
  k_sims_topk<<<dim3(28, 5, 25), 32, 0, stream>>>(FN, MASK, PARTIAL);
  k_reduce<<<1, 128, 0, stream>>>(PARTIAL, (float*)d_out);
}
